// SingleHeadAttention_69475390980772
// MI455X (gfx1250) — compile-verified, matrix operand order checked by reading, not run on hardware
//
#include <hip/hip_runtime.h>
#include <math.h>

// ---------------------------------------------------------------------------
// CDNA5 (gfx1250) single-head attention, bf16 WMMA pipeline + async-LDS DMA.
// Math identity (reference uses multiplicative tril mask + softmax over
// columns, so masked slots still contribute exp(0)=1 to the denominator):
//   colsum[j] = j + sum_{i>=j} exp(qk[i,j])
//   out[i]    = G + sum_{j<=i} (exp(qk[i,j]) - 1) * winv[j] * vp[j,:]
//   G[d]      = sum_j winv[j] * vp[j,d],   winv[j] = 1/colsum[j]
// qp/kp/vp kept in bf16 in workspace (24 MB -> resident in 192 MB L2).
// Heavy passes use GLOBAL_LOAD_ASYNC_TO_LDS_B128 double-buffered staging.
// ---------------------------------------------------------------------------

typedef __bf16 bf16;
typedef __attribute__((ext_vector_type(16))) __bf16 bf16x16;
typedef __attribute__((ext_vector_type(8)))  __bf16 bf16x8;
typedef __attribute__((ext_vector_type(8)))  float  v8f;

#define S_LEN   8192
#define D_MODEL 1024
#define DK      512
#define RSCALE  0.044194173824159216f   // 1/sqrt(512)

#define TILE    128
#define KSTEP   32
#define LDSS    40     // padded LDS stride (bf16) for 128x32 staging tiles
#define PSTR2   72     // padded LDS stride (bf16) for 128x64 P half-tile
#define NTHREADS 256   // 8 waves of 32

// ---- CDNA5 primitives ------------------------------------------------------

__device__ __forceinline__ v8f wmma_bf16(bf16x16 a, bf16x16 b, v8f c) {
  // D = A(16x32 bf16) x B(32x16 bf16) + C(16x16 f32)
  return __builtin_amdgcn_wmma_f32_16x16x32_bf16(
      /*neg_a=*/false, a, /*neg_b=*/false, b,
      /*c_mod=*/(short)0, c, /*reuse_a=*/false, /*reuse_b=*/false);
}

// Low 32 bits of a generic LDS address are the LDS byte offset (ISA 10.2).
__device__ __forceinline__ unsigned lds_off(const void* p) {
  return (unsigned)(size_t)p;
}

#define WAIT_ASYNC(n) asm volatile("s_wait_asynccnt %0" ::"i"(n) : "memory")

// Issue async global->LDS copies for one 128x32 bf16 tile pair (A and B).
// 256 threads: thread covers row tid>>1, 16 cols at ((tid&1)*16) => 2 x b128
// per tile per thread = 4 async instructions per wave per call (ASYNCcnt +=4).
__device__ __forceinline__ void stage_async_pair(bf16* dA, const bf16* __restrict__ sAg,
                                                 bf16* dB, const bf16* __restrict__ sBg,
                                                 int tid) {
  int r = tid >> 1;
  int c = (tid & 1) << 4;
  unsigned long long ga = (unsigned long long)(const void*)(sAg + r * DK + c);
  unsigned long long gb = (unsigned long long)(const void*)(sBg + r * DK + c);
  unsigned la = lds_off(dA + r * LDSS + c);
  unsigned lb = lds_off(dB + r * LDSS + c);
  asm volatile(
      "global_load_async_to_lds_b128 %0, %2, off\n\t"
      "global_load_async_to_lds_b128 %1, %3, off\n\t"
      "global_load_async_to_lds_b128 %4, %6, off\n\t"
      "global_load_async_to_lds_b128 %5, %7, off"
      :
      : "v"(la), "v"(la + 16u), "v"(ga), "v"(ga + 16ull),
        "v"(lb), "v"(lb + 16u), "v"(gb), "v"(gb + 16ull)
      : "memory");
}

// Load one 16x32 bf16 fragment from LDS (A layout; also serves as B when the
// staged tile is [N][K]).  Per ISA 7.12.2:
//   lane L: row = row0 + L%16, K base = (L/16)*8, elems {k..k+7, k+16..k+23}
__device__ __forceinline__ bf16x16 frag_ld(const bf16* base, int row0, int stride, int lane) {
  int r  = row0 + (lane & 15);
  int k0 = (lane >> 4) << 3;
  const bf16* p = base + r * stride + k0;
  union { bf16x16 v; bf16x8 h[2]; } u;
  u.h[0] = *(const bf16x8*)(p);
  u.h[1] = *(const bf16x8*)(p + 16);
  return u.v;
}

// Stage a 128x32 fp32 tile from global into LDS as bf16 (row stride LDSS).
__device__ __forceinline__ void stage_f32(bf16* dst, const float* __restrict__ src,
                                          int src_ld, int tid) {
  int r = tid >> 1;
  int c = (tid & 1) << 4;
  const float* p = src + r * src_ld + c;
  float4 f0 = *(const float4*)(p + 0);
  float4 f1 = *(const float4*)(p + 4);
  float4 f2 = *(const float4*)(p + 8);
  float4 f3 = *(const float4*)(p + 12);
  bf16* q = dst + r * LDSS + c;
  q[0]=(bf16)f0.x; q[1]=(bf16)f0.y; q[2]=(bf16)f0.z; q[3]=(bf16)f0.w;
  q[4]=(bf16)f1.x; q[5]=(bf16)f1.y; q[6]=(bf16)f1.z; q[7]=(bf16)f1.w;
  q[8]=(bf16)f2.x; q[9]=(bf16)f2.y; q[10]=(bf16)f2.z; q[11]=(bf16)f2.w;
  q[12]=(bf16)f3.x; q[13]=(bf16)f3.y; q[14]=(bf16)f3.z; q[15]=(bf16)f3.w;
}

// ---- kernel 0: colsum[j] = j (counts the exp(0)=1 masked contributions) ----

__global__ void init_colsum_kernel(float* __restrict__ colsum) {
  int i = blockIdx.x * blockDim.x + threadIdx.x;
  if (i < S_LEN) colsum[i] = (float)i;
}

// ---- kernel 1: projection GEMM  P = X @ W^T + b  (fp32 in, bf16 out) -------

__global__ __launch_bounds__(NTHREADS)
void proj_kernel(const float* __restrict__ X, const float* __restrict__ W,
                 const float* __restrict__ bias, bf16* __restrict__ out) {
  __shared__ __align__(16) bf16 sA[TILE * LDSS];
  __shared__ __align__(16) bf16 sB[TILE * LDSS];
  int tid = threadIdx.x, lane = tid & 31, wid = tid >> 5;
  int wm = wid & 3, wn = wid >> 2;               // 4x2 wave grid, 32x64 per wave
  int ib = blockIdx.x * TILE;
  int nb = blockIdx.y * TILE;

  v8f acc[2][4];
  #pragma unroll
  for (int i = 0; i < 2; ++i)
    #pragma unroll
    for (int j = 0; j < 4; ++j) acc[i][j] = (v8f)(0.0f);

  for (int k = 0; k < D_MODEL; k += KSTEP) {
    __syncthreads();
    stage_f32(sA, X + (size_t)ib * D_MODEL + k, D_MODEL, tid);
    stage_f32(sB, W + (size_t)nb * D_MODEL + k, D_MODEL, tid);
    __syncthreads();
    bf16x16 af[2], bfr[4];
    #pragma unroll
    for (int fi = 0; fi < 2; ++fi) af[fi] = frag_ld(sA, wm * 32 + fi * 16, LDSS, lane);
    #pragma unroll
    for (int fj = 0; fj < 4; ++fj) bfr[fj] = frag_ld(sB, wn * 64 + fj * 16, LDSS, lane);
    #pragma unroll
    for (int fi = 0; fi < 2; ++fi)
      #pragma unroll
      for (int fj = 0; fj < 4; ++fj)
        acc[fi][fj] = wmma_bf16(af[fi], bfr[fj], acc[fi][fj]);
  }

  #pragma unroll
  for (int fi = 0; fi < 2; ++fi)
    #pragma unroll
    for (int fj = 0; fj < 4; ++fj) {
      int n  = nb + wn * 64 + fj * 16 + (lane & 15);
      float b = bias[n];
      int m0 = ib + wm * 32 + fi * 16 + ((lane >> 4) << 3);
      #pragma unroll
      for (int r = 0; r < 8; ++r)
        out[(size_t)(m0 + r) * DK + n] = (bf16)(acc[fi][fj][r] + b);
    }
}

// ---- kernel 2: triangular colsum pass  colsum[j] += sum_{i>=j} exp(qk) -----

__global__ __launch_bounds__(NTHREADS)
void colsum_kernel(const bf16* __restrict__ qp, const bf16* __restrict__ kp,
                   float* __restrict__ colsum) {
  __shared__ __align__(16) bf16 sA[2][TILE * LDSS];
  __shared__ __align__(16) bf16 sB[2][TILE * LDSS];
  __shared__ float cbuf[TILE];
  int tid = threadIdx.x, lane = tid & 31, wid = tid >> 5;
  int wm = wid & 3, wn = wid >> 2;

  // decode triangular block index -> (it, jt) with jt <= it
  int bid = blockIdx.x;
  int it = (int)((sqrtf(8.0f * (float)bid + 1.0f) - 1.0f) * 0.5f);
  while ((it + 1) * (it + 2) / 2 <= bid) ++it;
  while (it * (it + 1) / 2 > bid) --it;
  int jt = bid - it * (it + 1) / 2;
  int ib = it * TILE, jb = jt * TILE;

  if (tid < TILE) cbuf[tid] = 0.0f;

  v8f acc[2][4];
  #pragma unroll
  for (int i = 0; i < 2; ++i)
    #pragma unroll
    for (int j = 0; j < 4; ++j) acc[i][j] = (v8f)(0.0f);

  const bf16* qrow = qp + (size_t)ib * DK;
  const bf16* krow = kp + (size_t)jb * DK;
  const int NK = DK / KSTEP;                       // 16
  stage_async_pair(sA[0], qrow, sB[0], krow, tid); // prologue: tile 0

  for (int ks = 0; ks < NK; ++ks) {
    int cur = ks & 1;
    if (ks + 1 < NK) {
      stage_async_pair(sA[cur ^ 1], qrow + (ks + 1) * KSTEP,
                       sB[cur ^ 1], krow + (ks + 1) * KSTEP, tid);
      WAIT_ASYNC(4);                               // current tile's 4 ops done
    } else {
      WAIT_ASYNC(0);
    }
    __syncthreads();                               // all waves' data visible
    bf16x16 af[2], bfr[4];
    #pragma unroll
    for (int fi = 0; fi < 2; ++fi) af[fi] = frag_ld(sA[cur], wm * 32 + fi * 16, LDSS, lane);
    #pragma unroll
    for (int fj = 0; fj < 4; ++fj) bfr[fj] = frag_ld(sB[cur], wn * 64 + fj * 16, LDSS, lane);
    #pragma unroll
    for (int fi = 0; fi < 2; ++fi)
      #pragma unroll
      for (int fj = 0; fj < 4; ++fj)
        acc[fi][fj] = wmma_bf16(af[fi], bfr[fj], acc[fi][fj]);
    __syncthreads();                               // guard buffer reuse
  }

  // exp + column reduce (mask i>=j)
  #pragma unroll
  for (int fi = 0; fi < 2; ++fi)
    #pragma unroll
    for (int fj = 0; fj < 4; ++fj) {
      int nl = wn * 64 + fj * 16 + (lane & 15);
      int gj = jb + nl;
      int m0 = ib + wm * 32 + fi * 16 + ((lane >> 4) << 3);
      float part = 0.0f;
      #pragma unroll
      for (int r = 0; r < 8; ++r) {
        int gi = m0 + r;
        if (gi >= gj) part += __expf(acc[fi][fj][r] * RSCALE);
      }
      part += __shfl_xor(part, 16);                // combine row-halves (same col)
      if (lane < 16) atomicAdd(&cbuf[nl], part);
    }
  __syncthreads();
  if (tid < TILE) atomicAdd(&colsum[jb + tid], cbuf[tid]);
}

// ---- kernel 3: winv[j] = 1/colsum[j] ---------------------------------------

__global__ void winv_kernel(const float* __restrict__ colsum, float* __restrict__ winv) {
  int i = blockIdx.x * blockDim.x + threadIdx.x;
  if (i < S_LEN) winv[i] = 1.0f / colsum[i];
}

// ---- kernel 4: G[d] = sum_j winv[j] * vp[j,d] ------------------------------

__global__ void gvec_kernel(const bf16* __restrict__ vp, const float* __restrict__ winv,
                            float* __restrict__ G) {
  int d = blockIdx.x * blockDim.x + threadIdx.x;
  if (d >= DK) return;
  float s = 0.0f;
  for (int j = 0; j < S_LEN; ++j) s += (float)vp[(size_t)j * DK + d] * winv[j];
  G[d] = s;
}

// ---- kernel 5: output pass -------------------------------------------------
//   out[i,d] = G[d] + sum_{jt<=it} [ (exp(qk)-1)*winv ]_tile @ vp_tile

__global__ __launch_bounds__(NTHREADS)
void out_kernel(const bf16* __restrict__ qp, const bf16* __restrict__ kp,
                const bf16* __restrict__ vp, const float* __restrict__ winv,
                const float* __restrict__ G, float* __restrict__ out) {
  __shared__ __align__(16) bf16 sA[2][TILE * LDSS];   // async staging / Vt chunk
  __shared__ __align__(16) bf16 sB[2][TILE * LDSS];
  __shared__ __align__(16) bf16 pT[TILE * PSTR2];     // 128x64 P half-tile
  int tid = threadIdx.x, lane = tid & 31, wid = tid >> 5;
  int wm = wid & 3, wn = wid >> 2;
  int it = blockIdx.x, dt = blockIdx.y;
  int ib = it * TILE, db = dt * TILE;

  v8f oacc[2][4];
  #pragma unroll
  for (int i = 0; i < 2; ++i)
    #pragma unroll
    for (int j = 0; j < 4; ++j) oacc[i][j] = (v8f)(0.0f);

  for (int jt = 0; jt <= it; ++jt) {
    int jb = jt * TILE;

    // --- phase A: S tile = qp_tile @ kp_tile^T (async double-buffered) -----
    v8f sacc[2][4];
    #pragma unroll
    for (int i = 0; i < 2; ++i)
      #pragma unroll
      for (int j = 0; j < 4; ++j) sacc[i][j] = (v8f)(0.0f);

    const bf16* qrow = qp + (size_t)ib * DK;
    const bf16* krow = kp + (size_t)jb * DK;
    const int NK = DK / KSTEP;                         // 16
    stage_async_pair(sA[0], qrow, sB[0], krow, tid);
    for (int ks = 0; ks < NK; ++ks) {
      int cur = ks & 1;
      if (ks + 1 < NK) {
        stage_async_pair(sA[cur ^ 1], qrow + (ks + 1) * KSTEP,
                         sB[cur ^ 1], krow + (ks + 1) * KSTEP, tid);
        WAIT_ASYNC(4);
      } else {
        WAIT_ASYNC(0);
      }
      __syncthreads();
      bf16x16 af[2], bfr[4];
      #pragma unroll
      for (int fi = 0; fi < 2; ++fi) af[fi] = frag_ld(sA[cur], wm * 32 + fi * 16, LDSS, lane);
      #pragma unroll
      for (int fj = 0; fj < 4; ++fj) bfr[fj] = frag_ld(sB[cur], wn * 64 + fj * 16, LDSS, lane);
      #pragma unroll
      for (int fi = 0; fi < 2; ++fi)
        #pragma unroll
        for (int fj = 0; fj < 4; ++fj)
          sacc[fi][fj] = wmma_bf16(af[fi], bfr[fj], sacc[fi][fj]);
      __syncthreads();
    }

    // --- phases B/C per 64-column half: P = (exp-1)*winv, then oacc += P@V --
    #pragma unroll
    for (int h = 0; h < 2; ++h) {
      if (wn == h) {                                  // this wave owns the half
        #pragma unroll
        for (int fi = 0; fi < 2; ++fi)
          #pragma unroll
          for (int fj = 0; fj < 4; ++fj) {
            int nl = fj * 16 + (lane & 15);           // 0..63 within half
            int gj = jb + h * 64 + nl;
            float w = winv[gj];
            int ml0 = wm * 32 + fi * 16 + ((lane >> 4) << 3);
            #pragma unroll
            for (int r = 0; r < 8; ++r) {
              int gi = ib + ml0 + r;
              float p = (gi >= gj) ? (__expf(sacc[fi][fj][r] * RSCALE) - 1.0f) * w : 0.0f;
              pT[(ml0 + r) * PSTR2 + nl] = (bf16)p;
            }
          }
      }
      __syncthreads();

      #pragma unroll
      for (int kk = 0; kk < 2; ++kk) {                // 2 K-steps of 32 local j
        {  // stage transposed V chunk: sA[0][d][j] <- vp[jb+h*64+kk*32+j][db+d]
          int d  = tid >> 1;
          int j0 = (tid & 1) << 4;
          const bf16* src = vp + (size_t)(jb + h * 64 + kk * 32 + j0) * DK + db + d;
          #pragma unroll
          for (int u = 0; u < 16; ++u)
            sA[0][d * LDSS + j0 + u] = src[(size_t)u * DK];
        }
        __syncthreads();
        bf16x16 af[2], bfr[4];
        #pragma unroll
        for (int fi = 0; fi < 2; ++fi)
          af[fi] = frag_ld(pT + kk * 32, wm * 32 + fi * 16, PSTR2, lane);
        #pragma unroll
        for (int fj = 0; fj < 4; ++fj)
          bfr[fj] = frag_ld(sA[0], wn * 64 + fj * 16, LDSS, lane);
        #pragma unroll
        for (int fi = 0; fi < 2; ++fi)
          #pragma unroll
          for (int fj = 0; fj < 4; ++fj)
            oacc[fi][fj] = wmma_bf16(af[fi], bfr[fj], oacc[fi][fj]);
        __syncthreads();
      }
    }
  }

  // epilogue: + G, fp32 store
  #pragma unroll
  for (int fi = 0; fi < 2; ++fi)
    #pragma unroll
    for (int fj = 0; fj < 4; ++fj) {
      int n = db + wn * 64 + fj * 16 + (lane & 15);
      float g = G[n];
      int m0 = ib + wm * 32 + fi * 16 + ((lane >> 4) << 3);
      #pragma unroll
      for (int r = 0; r < 8; ++r)
        out[(size_t)(m0 + r) * DK + n] = oacc[fi][fj][r] + g;
    }
}

// ---------------------------------------------------------------------------

extern "C" void kernel_launch(void* const* d_in, const int* in_sizes, int n_in,
                              void* d_out, int out_size, void* d_ws, size_t ws_size,
                              hipStream_t stream) {
  const float* q    = (const float*)d_in[0];
  const float* k    = (const float*)d_in[1];
  const float* v    = (const float*)d_in[2];
  const float* WQ_w = (const float*)d_in[3];
  const float* WQ_b = (const float*)d_in[4];
  const float* WK_w = (const float*)d_in[5];
  const float* WK_b = (const float*)d_in[6];
  const float* WV_w = (const float*)d_in[7];
  const float* WV_b = (const float*)d_in[8];
  float* out = (float*)d_out;

  // workspace layout (bytes): qp/kp/vp bf16 strips + stats (~25.2 MB total)
  const size_t PROJ_BYTES = (size_t)S_LEN * DK * sizeof(bf16);   // 8 MB each
  char* ws = (char*)d_ws;
  bf16*  qp     = (bf16*)(ws);
  bf16*  kp     = (bf16*)(ws + PROJ_BYTES);
  bf16*  vp     = (bf16*)(ws + 2 * PROJ_BYTES);
  float* colsum = (float*)(ws + 3 * PROJ_BYTES);
  float* winv   = (float*)(ws + 3 * PROJ_BYTES + S_LEN * sizeof(float));
  float* G      = (float*)(ws + 3 * PROJ_BYTES + 2 * S_LEN * sizeof(float));

  init_colsum_kernel<<<(S_LEN + 255) / 256, 256, 0, stream>>>(colsum);

  dim3 pgrid(S_LEN / TILE, DK / TILE);   // 64 x 4
  proj_kernel<<<pgrid, NTHREADS, 0, stream>>>(q, WQ_w, WQ_b, qp);
  proj_kernel<<<pgrid, NTHREADS, 0, stream>>>(k, WK_w, WK_b, kp);
  proj_kernel<<<pgrid, NTHREADS, 0, stream>>>(v, WV_w, WV_b, vp);

  int ntile = S_LEN / TILE;                           // 64
  int ntri  = ntile * (ntile + 1) / 2;                // 2080
  colsum_kernel<<<ntri, NTHREADS, 0, stream>>>(qp, kp, colsum);

  winv_kernel<<<(S_LEN + 255) / 256, 256, 0, stream>>>(colsum, winv);
  gvec_kernel<<<(DK + 255) / 256, 256, 0, stream>>>(vp, winv, G);

  dim3 ogrid(ntile, DK / TILE);          // 64 x 4
  out_kernel<<<ogrid, NTHREADS, 0, stream>>>(qp, kp, vp, winv, G, out);
}
